// MultiLayerGAT_65128884076840
// MI455X (gfx1250) — compile-verified, matrix operand order checked by reading, not run on hardware
//
#include <hip/hip_runtime.h>

// ---------------- problem constants ----------------
#define NN      4096
#define NFEAT   512
#define NHID    64
#define NCLASS  40
#define NHEADS  4
#define NEDGE   131072
#define H2DIM   (NHEADS * NHID)   // 256
#define NCPAD   48                // NCLASS padded to 3x16 WMMA tiles
#define ALPHA   0.2f
#define NEGV    -9e15f
#define ADJW    (NN / 32)         // 128 u32 words per adjacency row
#define JT      (NN / 32)         // 128 j-tiles of 32

typedef __attribute__((ext_vector_type(16))) __bf16 v16bf;
typedef __attribute__((ext_vector_type(8)))  float  v8f;

// ---------------- small helpers ----------------
static __device__ __forceinline__ unsigned short f2bfbits(float f) {
  union { __bf16 b; unsigned short u; } t;
  t.b = (__bf16)f;                      // hardware f32->bf16 cvt
  return t.u;
}
static __device__ __forceinline__ v8f wmma_bf16(v16bf a, v16bf b, v8f c) {
  // D = A(16x32 bf16) x B(32x16 bf16) + C(16x16 f32)
  return __builtin_amdgcn_wmma_f32_16x16x32_bf16(false, a, false, b, (short)0, c,
                                                 false, false);
}
// K coordinate of element e (0..15) of an A/B fragment (ISA 7.12.2, 16-bit 16x32):
// lanes 0-15: VGPR0-3 -> K=0..7, VGPR4-7 -> K=16..23 ; lanes 16-31: +8
static __device__ __forceinline__ int frag_k(int lane, int e) {
  int v = e >> 1, w = e & 1;
  int base = (v < 4) ? (2 * v + w) : (16 + 2 * (v - 4) + w);
  return base + ((lane >= 16) ? 8 : 0);
}
static __device__ __forceinline__ v16bf load_frag(const unsigned short* p) {
  return *(const v16bf*)p;              // 32B contiguous, 32B aligned
}
// low 32 bits of a generic LDS pointer = byte offset inside LDS
static __device__ __forceinline__ unsigned lds_off(const void* p) {
  return (unsigned)(unsigned long long)p;
}
// async copy of 32 contiguous bytes global -> LDS (2x B128, ASYNCcnt +2)
static __device__ __forceinline__ void async_copy32(unsigned lds, const void* g) {
  asm volatile("global_load_async_to_lds_b128 %0, %1, off\n\t"
               "global_load_async_to_lds_b128 %0, %1, off offset:16"
               :
               : "v"(lds), "v"((unsigned long long)g)
               : "memory");
}
static __device__ __forceinline__ void wait_async_le2() {
  asm volatile("s_wait_asynccnt 0x2" ::: "memory");
}
static __device__ __forceinline__ void wait_async_0() {
  asm volatile("s_wait_asynccnt 0x0" ::: "memory");
}

// ---------------- utility kernels ----------------
__global__ void zero_u32(unsigned* p, int n) {
  int i = blockIdx.x * blockDim.x + threadIdx.x;
  if (i < n) p[i] = 0u;
}

__global__ void build_adj(const long long* ei, unsigned* adj) {
  int e = blockIdx.x * blockDim.x + threadIdx.x;
  if (e >= NEDGE) return;
  int i = (int)ei[e];
  int j = (int)ei[NEDGE + e];
  atomicOr(&adj[(size_t)i * ADJW + (j >> 5)], 1u << (j & 31));
}

__global__ void cvt_f32_bf16(const float* in, unsigned short* out, int n) {
  int i = blockIdx.x * blockDim.x + threadIdx.x;
  if (i < n) out[i] = f2bfbits(in[i]);
}

// ---------------- B-side fragment packing ----------------
// Packed layout: [...tile...][lane(32)][e(16)] contiguous bf16, so a lane's
// whole fragment is one 32B vector load.

// W1 [NHEADS][NFEAT][NHID] f32 -> packed [h][kt(16)][t(4)][lane][e]
__global__ void pack_w1(const float* __restrict__ W1, unsigned short* __restrict__ out) {
  int idx = blockIdx.x * blockDim.x + threadIdx.x;
  if (idx >= NHEADS * 16 * 4 * 32 * 16) return;
  int e = idx & 15, lane = (idx >> 4) & 31, t = (idx >> 9) & 3;
  int kt = (idx >> 11) & 15, h = idx >> 15;
  int k = kt * 32 + frag_k(lane, e), n = t * 16 + (lane & 15);
  out[idx] = f2bfbits(W1[((size_t)h * NFEAT + k) * NHID + n]);
}

// Wh [NHEADS*NN][NHID] f32 -> packed [h][jt(128)][t(4)][lane][e]
__global__ void pack_wh(const float* __restrict__ wh, unsigned short* __restrict__ out) {
  int idx = blockIdx.x * blockDim.x + threadIdx.x;
  if (idx >= NHEADS * JT * 4 * 32 * 16) return;
  int e = idx & 15, lane = (idx >> 4) & 31, t = (idx >> 9) & 3;
  int jt = (idx >> 11) & 127, h = idx >> 18;
  int j = jt * 32 + frag_k(lane, e), n = t * 16 + (lane & 15);
  out[idx] = f2bfbits(wh[((size_t)h * NN + j) * NHID + n]);
}

// Wo [H2DIM][NCLASS] f32 -> packed [kt(8)][t(3)][lane][e] (cols padded to 48)
// also builds padded bias
__global__ void pack_wo(const float* __restrict__ Wo, const float* __restrict__ bo,
                        unsigned short* __restrict__ out, float* __restrict__ bob) {
  int idx = blockIdx.x * blockDim.x + threadIdx.x;
  if (idx < 8 * 3 * 32 * 16) {
    int e = idx & 15, lane = (idx >> 4) & 31;
    int t = (idx >> 9) % 3, kt = (idx >> 9) / 3;
    int k = kt * 32 + frag_k(lane, e), n = t * 16 + (lane & 15);
    float v = (n < NCLASS) ? Wo[(size_t)k * NCLASS + n] : 0.f;
    out[idx] = f2bfbits(v);
  }
  if (idx < NCPAD) bob[idx] = (idx < NCLASS) ? bo[idx] : 0.f;
}

// Who [NN][NCPAD] f32 (pad cols already 0) -> packed [jt(128)][t(3)][lane][e]
__global__ void pack_who(const float* __restrict__ who, unsigned short* __restrict__ out) {
  int idx = blockIdx.x * blockDim.x + threadIdx.x;
  if (idx >= JT * 3 * 32 * 16) return;
  int e = idx & 15, lane = (idx >> 4) & 31;
  int t = (idx >> 9) % 3, jt = (idx >> 9) / 3;
  int j = jt * 32 + frag_k(lane, e), n = t * 16 + (lane & 15);
  out[idx] = f2bfbits(who[(size_t)j * NCPAD + n]);
}

// ---------------- layer-1 projection: Wh[h] = X @ W1[h] + b1[h] ----------------
// block = 256 (8 waves); wave -> 16 rows x 64 cols; grid = (32, NHEADS)
__global__ __launch_bounds__(256) void gemm1_wmma(
    const unsigned short* __restrict__ xb, const unsigned short* __restrict__ w1p,
    const float* __restrict__ b1, float* __restrict__ wh) {
  int lane = threadIdx.x & 31, wv = threadIdx.x >> 5;
  int h = blockIdx.y;
  int row0 = (blockIdx.x * 8 + wv) * 16;
  int M = lane & 15;
  int half8 = (lane >= 16) ? 8 : 0;
  v8f z = {};
  v8f acc[4] = {z, z, z, z};

  const unsigned short* bp = w1p + ((size_t)h * 16 * 4) * 32 * 16 + lane * 16;
  for (int k0 = 0; k0 < NFEAT; k0 += 32) {
    v16bf A;
#pragma unroll
    for (int e = 0; e < 16; ++e)
      A[e] = ((const __bf16*)xb)[(size_t)(row0 + M) * NFEAT + k0 + frag_k(lane, e)];
#pragma unroll
    for (int t = 0; t < 4; ++t) {
      v16bf B = load_frag(bp + (size_t)t * 32 * 16);
      acc[t] = wmma_bf16(A, B, acc[t]);
    }
    bp += 4 * 32 * 16;
  }
#pragma unroll
  for (int t = 0; t < 4; ++t)
#pragma unroll
    for (int v = 0; v < 8; ++v) {
      int mr = v + half8, nc = t * 16 + M;
      wh[((size_t)h * NN + row0 + mr) * NHID + nc] = acc[t][v] + b1[h * NHID + nc];
    }
}

// src[h][i] = Wh[h][i,:].a1[h][:64]; dst[h][i] = Wh[h][i,:].a1[h][64:]
__global__ void srcdst1(const float* __restrict__ wh, const float* __restrict__ a1,
                        float* __restrict__ src, float* __restrict__ dst) {
  int idx = blockIdx.x * blockDim.x + threadIdx.x;
  if (idx >= NHEADS * NN) return;
  int h = idx / NN;
  const float* row = wh + (size_t)idx * NHID;
  float s = 0.f, d = 0.f;
  for (int f = 0; f < NHID; ++f) {
    s += row[f] * a1[h * 2 * NHID + f];
    d += row[f] * a1[h * 2 * NHID + NHID + f];
  }
  src[idx] = s;
  dst[idx] = d;
}

// ---------------- fused flash GAT, layer 1 ----------------
// Block = 4 waves, all on the same head and the same j-stream; each wave owns a
// 16-row tile. The packed V tile (4KB) is async-copied to LDS once per block
// (double buffered, ASYNCcnt pipelined), cutting L2 read traffic 4x.
__global__ __launch_bounds__(128) void flash_gat1(
    const unsigned* __restrict__ adj, const unsigned short* __restrict__ whp,
    const float* __restrict__ src, const float* __restrict__ dst,
    const float* __restrict__ ab1, float* __restrict__ h2,
    unsigned short* __restrict__ h2b) {
  __shared__ __attribute__((aligned(32))) unsigned short vtile[2][4 * 32 * 16];
  int tid = threadIdx.x;
  int lane = tid & 31, wv = tid >> 5;
  int h = blockIdx.y;
  int i0 = blockIdx.x * 64 + wv * 16;
  int M = lane & 15;
  int half8 = (lane >= 16) ? 8 : 0;
  float ab = ab1[h];
  float srM = src[h * NN + i0 + M];
  const unsigned* adjrow = adj + (size_t)(i0 + M) * ADJW;
  const unsigned short* vbase = whp + ((size_t)h * JT * 4) * 32 * 16;

  float m = -INFINITY, l = 0.f;
  v8f z = {};
  v8f acc[4] = {z, z, z, z};

  // prologue: stage tile 0 (each thread: 32B)
  async_copy32(lds_off(&vtile[0][tid * 16]), vbase + tid * 16);

  for (int jt = 0; jt < JT; ++jt) {
    int buf = jt & 1;
    if (jt + 1 < JT) {   // stage next tile into the other buffer
      async_copy32(lds_off(&vtile[buf ^ 1][tid * 16]),
                   vbase + (size_t)(jt + 1) * 4 * 32 * 16 + tid * 16);
      wait_async_le2();  // in-order: the 2 older loads (current buf) are done
    } else {
      wait_async_0();
    }
    __syncthreads();

    int j0 = jt * 32;
    float d = dst[h * NN + j0 + lane];
    unsigned aw = adjrow[jt];
    float ev[16];
    float mx = -INFINITY;
#pragma unroll
    for (int e = 0; e < 16; ++e) {
      int K = frag_k(lane, e);
      float dk = __shfl(d, K, 32);
      float t = srM + dk + ab;
      t = (t > 0.f) ? t : ALPHA * t;              // leaky_relu before mask
      t = ((aw >> K) & 1u) ? t : NEGV;            // adjacency mask
      ev[e] = t;
      mx = fmaxf(mx, t);
    }
    mx = fmaxf(mx, __shfl_xor(mx, 16, 32));       // full-row max
    float mn = fmaxf(m, mx);
    float sc = __expf(m - mn);
    float s = 0.f;
    v16bf A;
#pragma unroll
    for (int e = 0; e < 16; ++e) {
      float p = __expf(ev[e] - mn);
      s += p;
      A[e] = (__bf16)p;
    }
    s += __shfl_xor(s, 16, 32);                   // full-row sum
    l = l * sc + s;
    m = mn;
    float scv[8];
#pragma unroll
    for (int v = 0; v < 8; ++v) scv[v] = __shfl(sc, v + half8, 32);
#pragma unroll
    for (int t = 0; t < 4; ++t) {
#pragma unroll
      for (int v = 0; v < 8; ++v) acc[t][v] *= scv[v];
      v16bf B = load_frag(&vtile[buf][(t * 32 + lane) * 16]);  // ds_load_b128 x2
      acc[t] = wmma_bf16(A, B, acc[t]);
    }
    __syncthreads();   // everyone done reading buf before it is restaged
  }
  float linv = 1.f / l;
#pragma unroll
  for (int t = 0; t < 4; ++t)
#pragma unroll
    for (int v = 0; v < 8; ++v) {
      int mr = v + half8;
      float val = acc[t][v] * __shfl(linv, mr, 32);
      val = (val > 0.f) ? val : (__expf(val) - 1.f);   // elu
      size_t idx = (size_t)(i0 + mr) * H2DIM + h * NHID + t * 16 + M;
      h2[idx] = val;
      h2b[idx] = f2bfbits(val);
    }
}

// ---------------- output projection: Who = h2 @ Wo + bo (cols padded to 48) ----
__global__ __launch_bounds__(256) void gemm_out_wmma(
    const unsigned short* __restrict__ h2b, const unsigned short* __restrict__ wop,
    const float* __restrict__ bob, float* __restrict__ who) {
  int lane = threadIdx.x & 31, wv = threadIdx.x >> 5;
  int row0 = (blockIdx.x * 8 + wv) * 16;
  int M = lane & 15;
  int half8 = (lane >= 16) ? 8 : 0;
  v8f z = {};
  v8f acc[3] = {z, z, z};
  const unsigned short* bp = wop + lane * 16;
  for (int k0 = 0; k0 < H2DIM; k0 += 32) {
    v16bf A;
#pragma unroll
    for (int e = 0; e < 16; ++e)
      A[e] = ((const __bf16*)h2b)[(size_t)(row0 + M) * H2DIM + k0 + frag_k(lane, e)];
#pragma unroll
    for (int t = 0; t < 3; ++t) {
      v16bf B = load_frag(bp + (size_t)t * 32 * 16);
      acc[t] = wmma_bf16(A, B, acc[t]);
    }
    bp += 3 * 32 * 16;
  }
#pragma unroll
  for (int t = 0; t < 3; ++t)
#pragma unroll
    for (int v = 0; v < 8; ++v) {
      int mr = v + half8, nc = t * 16 + M;
      who[(size_t)(row0 + mr) * NCPAD + nc] = acc[t][v] + bob[nc];
    }
}

__global__ void srcdst_out(const float* __restrict__ who, const float* __restrict__ ao,
                           float* __restrict__ src, float* __restrict__ dst) {
  int i = blockIdx.x * blockDim.x + threadIdx.x;
  if (i >= NN) return;
  const float* row = who + (size_t)i * NCPAD;
  float s = 0.f, d = 0.f;
  for (int f = 0; f < NCLASS; ++f) {
    s += row[f] * ao[f];
    d += row[f] * ao[NCLASS + f];
  }
  src[i] = s;
  dst[i] = d;
}

// ---------------- fused flash GAT output layer + elu + log_softmax ----------------
__global__ __launch_bounds__(128) void flash_gat_out(
    const unsigned* __restrict__ adj, const unsigned short* __restrict__ whop,
    const float* __restrict__ src, const float* __restrict__ dst,
    const float* __restrict__ abo, float* __restrict__ out) {
  __shared__ __attribute__((aligned(32))) unsigned short vtile[2][3 * 32 * 16];
  __shared__ float stage[4][16][NCPAD + 1];   // final log_softmax staging
  int tid = threadIdx.x;
  int lane = tid & 31, wv = tid >> 5;
  int i0 = blockIdx.x * 64 + wv * 16;
  int M = lane & 15;
  int half8 = (lane >= 16) ? 8 : 0;
  float ab = abo[0];
  float srM = src[i0 + M];
  const unsigned* adjrow = adj + (size_t)(i0 + M) * ADJW;

  float m = -INFINITY, l = 0.f;
  v8f z = {};
  v8f acc[3] = {z, z, z};

  // tile is 3KB = 96 x 32B chunks; threads 96..127 duplicate chunks 64..95
  // (same src -> same dst, idempotent) so EXEC and ASYNCcnt stay uniform.
  int chunk = (tid < 96) ? tid : (tid - 32);
  async_copy32(lds_off(&vtile[0][chunk * 16]), whop + chunk * 16);

  for (int jt = 0; jt < JT; ++jt) {
    int buf = jt & 1;
    if (jt + 1 < JT) {
      async_copy32(lds_off(&vtile[buf ^ 1][chunk * 16]),
                   whop + (size_t)(jt + 1) * 3 * 32 * 16 + chunk * 16);
      wait_async_le2();
    } else {
      wait_async_0();
    }
    __syncthreads();

    int j0 = jt * 32;
    float d = dst[j0 + lane];
    unsigned aw = adjrow[jt];
    float ev[16];
    float mx = -INFINITY;
#pragma unroll
    for (int e = 0; e < 16; ++e) {
      int K = frag_k(lane, e);
      float dk = __shfl(d, K, 32);
      float t = srM + dk + ab;
      t = (t > 0.f) ? t : ALPHA * t;
      t = ((aw >> K) & 1u) ? t : NEGV;
      ev[e] = t;
      mx = fmaxf(mx, t);
    }
    mx = fmaxf(mx, __shfl_xor(mx, 16, 32));
    float mn = fmaxf(m, mx);
    float sc = __expf(m - mn);
    float s = 0.f;
    v16bf A;
#pragma unroll
    for (int e = 0; e < 16; ++e) {
      float p = __expf(ev[e] - mn);
      s += p;
      A[e] = (__bf16)p;
    }
    s += __shfl_xor(s, 16, 32);
    l = l * sc + s;
    m = mn;
    float scv[8];
#pragma unroll
    for (int v = 0; v < 8; ++v) scv[v] = __shfl(sc, v + half8, 32);
#pragma unroll
    for (int t = 0; t < 3; ++t) {
#pragma unroll
      for (int v = 0; v < 8; ++v) acc[t][v] *= scv[v];
      v16bf B = load_frag(&vtile[buf][(t * 32 + lane) * 16]);
      acc[t] = wmma_bf16(A, B, acc[t]);
    }
    __syncthreads();
  }
  // normalize + elu -> stage rows in LDS for log_softmax
  float linv = 1.f / l;
#pragma unroll
  for (int t = 0; t < 3; ++t)
#pragma unroll
    for (int v = 0; v < 8; ++v) {
      int mr = v + half8;
      float val = acc[t][v] * __shfl(linv, mr, 32);
      val = (val > 0.f) ? val : (__expf(val) - 1.f);
      stage[wv][mr][t * 16 + M] = val;
    }
  __syncthreads();
  if (lane < 16) {
    int r = lane;
    float mx = -INFINITY;
    for (int c = 0; c < NCLASS; ++c) mx = fmaxf(mx, stage[wv][r][c]);
    float s = 0.f;
    for (int c = 0; c < NCLASS; ++c) s += __expf(stage[wv][r][c] - mx);
    float ls = __logf(s);
    for (int c = 0; c < NCLASS; ++c)
      out[(size_t)(i0 + r) * NCLASS + c] = stage[wv][r][c] - mx - ls;
  }
}

// ---------------- host-side launcher ----------------
extern "C" void kernel_launch(void* const* d_in, const int* in_sizes, int n_in,
                              void* d_out, int out_size, void* d_ws, size_t ws_size,
                              hipStream_t stream) {
  (void)in_sizes; (void)n_in; (void)out_size; (void)ws_size;
  const float*     x    = (const float*)d_in[0];
  const long long* ei   = (const long long*)d_in[1];   // int64 edge_index [2,E]
  const float*     W1   = (const float*)d_in[2];
  const float*     b1   = (const float*)d_in[3];
  const float*     a1   = (const float*)d_in[4];
  const float*     ab1  = (const float*)d_in[5];
  const float*     Wo   = (const float*)d_in[6];
  const float*     bo   = (const float*)d_in[7];
  const float*     ao   = (const float*)d_in[8];
  const float*     abo  = (const float*)d_in[9];

  char* ws = (char*)d_ws;
  size_t off = 0;
  auto take = [&](size_t bytes) {
    void* p = ws + off;
    off = (off + bytes + 255) & ~(size_t)255;
    return p;
  };
  unsigned*       adj  = (unsigned*)      take((size_t)NN * ADJW * 4);
  unsigned short* xb   = (unsigned short*)take((size_t)NN * NFEAT * 2);
  unsigned short* w1p  = (unsigned short*)take((size_t)NHEADS * 16 * 4 * 32 * 16 * 2);
  float*          wh   = (float*)         take((size_t)NHEADS * NN * NHID * 4);
  unsigned short* whp  = (unsigned short*)take((size_t)NHEADS * JT * 4 * 32 * 16 * 2);
  float*          src1 = (float*)         take((size_t)NHEADS * NN * 4);
  float*          dst1 = (float*)         take((size_t)NHEADS * NN * 4);
  float*          h2   = (float*)         take((size_t)NN * H2DIM * 4);
  unsigned short* h2b  = (unsigned short*)take((size_t)NN * H2DIM * 2);
  unsigned short* wop  = (unsigned short*)take((size_t)8 * 3 * 32 * 16 * 2);
  float*          bob  = (float*)         take((size_t)NCPAD * 4);
  float*          who  = (float*)         take((size_t)NN * NCPAD * 4);
  unsigned short* whop = (unsigned short*)take((size_t)JT * 3 * 32 * 16 * 2);
  float*          srco = (float*)         take((size_t)NN * 4);
  float*          dsto = (float*)         take((size_t)NN * 4);

  // adjacency bitmask
  zero_u32<<<(NN * ADJW + 255) / 256, 256, 0, stream>>>(adj, NN * ADJW);
  build_adj<<<(NEDGE + 255) / 256, 256, 0, stream>>>(ei, adj);
  // operand conversion / packing
  cvt_f32_bf16<<<(NN * NFEAT + 255) / 256, 256, 0, stream>>>(x, xb, NN * NFEAT);
  pack_w1<<<(NHEADS * 16 * 4 * 32 * 16 + 255) / 256, 256, 0, stream>>>(W1, w1p);
  // layer 1
  gemm1_wmma<<<dim3(32, NHEADS), 256, 0, stream>>>(xb, w1p, b1, wh);
  pack_wh<<<(NHEADS * JT * 4 * 32 * 16 + 255) / 256, 256, 0, stream>>>(wh, whp);
  srcdst1<<<(NHEADS * NN + 255) / 256, 256, 0, stream>>>(wh, a1, src1, dst1);
  flash_gat1<<<dim3(NN / 64, NHEADS), 128, 0, stream>>>(adj, whp, src1, dst1,
                                                        ab1, h2, h2b);
  // output layer
  pack_wo<<<(8 * 3 * 32 * 16 + 255) / 256, 256, 0, stream>>>(Wo, bo, wop, bob);
  gemm_out_wmma<<<32, 256, 0, stream>>>(h2b, wop, bob, who);
  pack_who<<<(JT * 3 * 32 * 16 + 255) / 256, 256, 0, stream>>>(who, whop);
  srcdst_out<<<(NN + 255) / 256, 256, 0, stream>>>(who, ao, srco, dsto);
  flash_gat_out<<<NN / 64, 128, 0, stream>>>(adj, whop, srco, dsto, abo,
                                             (float*)d_out);
}
